// MultiheadAttention_2680059593303
// MI455X (gfx1250) — matrix operand order repaired and checked by reading, not compiled, not run on hardware
//
#include <hip/hip_runtime.h>
#include <hip/hip_bf16.h>

typedef __attribute__((ext_vector_type(16))) _Float16 v16h;
typedef __attribute__((ext_vector_type(8)))  float    v8f;

union HV { v16h h; unsigned int u[8]; };

#define WMMA_F16(A, B, C) \
  __builtin_amdgcn_wmma_f32_16x16x32_f16(false, (A), false, (B), (short)0, (C), false, false)

// ---- WMMA operand loaders (wave32, 16x16x32 f16) -------------------------
// A (16x32, MxK): lane m = lane&15, group g = lane>>4.
//   pair v: k = 2v + 16g -> dword v + 8g  (2x b128)
__device__ __forceinline__ v16h load_a_f16(const _Float16* __restrict__ row, int g) {
  HV a;
  const unsigned int* p = (const unsigned int*)row;
#pragma unroll
  for (int v = 0; v < 8; ++v) a.u[v] = p[v + 8 * g];
  return a.h;
}
// B (32x16, KxN): lane n = lane&15; column n is a contiguous K-vector.
//   pair v: k = 2v + 16g -> dword v + 8g  (2x b128)
__device__ __forceinline__ v16h load_b_f16(const _Float16* __restrict__ col, int g) {
  HV b;
  const unsigned int* p = (const unsigned int*)col;
#pragma unroll
  for (int v = 0; v < 8; ++v) b.u[v] = p[v + 8 * g];
  return b.h;
}

// ---- Kernel 0: one-time fp32 -> f16 conversion (vectorized) --------------
__global__ __launch_bounds__(256) void k_cvt(const float* __restrict__ src,
                                             _Float16* __restrict__ dst, int n4) {
  int i = blockIdx.x * 256 + threadIdx.x;
  if (i >= n4) return;
  float4 v = ((const float4*)src)[i];
  union { _Float16 h[4]; unsigned long long q; } u;
  u.h[0] = (_Float16)v.x; u.h[1] = (_Float16)v.y;
  u.h[2] = (_Float16)v.z; u.h[3] = (_Float16)v.w;
  ((unsigned long long*)dst)[i] = u.q;
}

// ---- Kernel 1: fused Q/K/V projections  P = f16(Xh @ Wh^T + b) -----------
// 12288 waves: 3 matrices x 256 row-tiles x 16 n-blocks(64).
// Software-pipelined: next K-chunk operands prefetched before current WMMAs.
__global__ __launch_bounds__(128) void k_proj(
    const _Float16* __restrict__ Xh,
    const _Float16* __restrict__ Wq, const float* __restrict__ bq, _Float16* __restrict__ Pq,
    const _Float16* __restrict__ Wk, const float* __restrict__ bk, _Float16* __restrict__ Pk,
    const _Float16* __restrict__ Wv, const float* __restrict__ bv, _Float16* __restrict__ Pv)
{
  const int N = 1024, K = 1024;
  int wid   = blockIdx.x * 4 + (threadIdx.x >> 5);
  int which = wid / 4096;
  int t     = wid - which * 4096;
  int tm    = (t >> 4) << 4;
  int tn    = (t & 15) << 6;
  const _Float16* W    = which == 0 ? Wq : (which == 1 ? Wk : Wv);
  const float*    bias = which == 0 ? bq : (which == 1 ? bk : bv);
  _Float16*       P    = which == 0 ? Pq : (which == 1 ? Pk : Pv);

  int lane = threadIdx.x & 31, g = lane >> 4, ln = lane & 15;
  v8f acc0 = {}, acc1 = {}, acc2 = {}, acc3 = {};
  const _Float16* arow = Xh + (size_t)(tm + ln) * K;
  const _Float16* wrow = W  + (size_t)(tn + ln) * K;

  v16h a  = load_a_f16(arow, g);
  v16h b0 = load_b_f16(wrow, g);
  v16h b1 = load_b_f16(wrow + (size_t)16 * K, g);
  v16h b2 = load_b_f16(wrow + (size_t)32 * K, g);
  v16h b3 = load_b_f16(wrow + (size_t)48 * K, g);
  for (int k0 = 32; k0 < K; k0 += 32) {
    v16h an  = load_a_f16(arow + k0, g);
    v16h bn0 = load_b_f16(wrow + k0, g);
    v16h bn1 = load_b_f16(wrow + (size_t)16 * K + k0, g);
    v16h bn2 = load_b_f16(wrow + (size_t)32 * K + k0, g);
    v16h bn3 = load_b_f16(wrow + (size_t)48 * K + k0, g);
    acc0 = WMMA_F16(a, b0, acc0);
    acc1 = WMMA_F16(a, b1, acc1);
    acc2 = WMMA_F16(a, b2, acc2);
    acc3 = WMMA_F16(a, b3, acc3);
    a = an; b0 = bn0; b1 = bn1; b2 = bn2; b3 = bn3;
  }
  acc0 = WMMA_F16(a, b0, acc0);
  acc1 = WMMA_F16(a, b1, acc1);
  acc2 = WMMA_F16(a, b2, acc2);
  acc3 = WMMA_F16(a, b3, acc3);

  float bb0 = bias[tn + ln], bb1 = bias[tn + 16 + ln];
  float bb2 = bias[tn + 32 + ln], bb3 = bias[tn + 48 + ln];
#pragma unroll
  for (int j = 0; j < 8; ++j) {
    size_t base = (size_t)(tm + j + 8 * g) * N + tn + ln;
    P[base]      = (_Float16)(acc0[j] + bb0);
    P[base + 16] = (_Float16)(acc1[j] + bb1);
    P[base + 32] = (_Float16)(acc2[j] + bb2);
    P[base + 48] = (_Float16)(acc3[j] + bb3);
  }
}

// ---- Kernel 2: per-head V transpose  Vt[head][hh][s] = Pv[head][s][hh] ---
__global__ __launch_bounds__(256) void k_vt(const _Float16* __restrict__ Pv,
                                            _Float16* __restrict__ Vt) {
  int i = blockIdx.x * 256 + threadIdx.x;  // 2M outputs of 2 halfs each
  int s2   = (i & 1023) << 1;
  int hh   = (i >> 10) & 31;
  int head = i >> 15;
  const _Float16* p = Pv + (size_t)head * 65536 + (size_t)s2 * 32 + hh;
  size_t o = (size_t)head * 65536 + (size_t)hh * 2048 + s2;
  Vt[o]     = p[0];
  Vt[o + 1] = p[32];
}

// ---- Kernel 3: fused per-head attention (no softmax path) ----------------
__device__ __forceinline__ void attn_step(
    v16h aq, v16h kb0, v16h kb1, v16h bv0, v16h bv1,
    _Float16 (* __restrict__ S)[32], int g, int ln, v8f& acc0, v8f& acc1)
{
  v8f sc0 = {}, sc1 = {};
  sc0 = WMMA_F16(aq, kb0, sc0);
  sc1 = WMMA_F16(aq, kb1, sc1);
#pragma unroll
  for (int j = 0; j < 8; ++j) {
    S[j + 8 * g][ln]      = (_Float16)sc0[j];
    S[j + 8 * g][16 + ln] = (_Float16)sc1[j];
  }
  v16h as = load_a_f16(&S[ln][0], g);
  acc0 = WMMA_F16(as, bv0, acc0);
  acc1 = WMMA_F16(as, bv1, acc1);
}

// A2[head] (2048x32) = (Q_h K_h^T) V_h, scores live only in regs/LDS.
// 8192 waves: 64 heads x 128 t-tiles. Pipelined over s-chunks.
__global__ __launch_bounds__(128) void k_attn(
    const _Float16* __restrict__ Pq, const _Float16* __restrict__ Pk,
    const _Float16* __restrict__ Vt, _Float16* __restrict__ A2)
{
  int wid  = blockIdx.x * 4 + (threadIdx.x >> 5);
  int head = wid >> 7;
  int t0   = (wid & 127) << 4;
  const _Float16* Qh  = Pq + (size_t)head * 65536;
  const _Float16* Kh  = Pk + (size_t)head * 65536;
  const _Float16* Vth = Vt + (size_t)head * 65536;
  int lane = threadIdx.x & 31, g = lane >> 4, ln = lane & 15;
  int ws   = threadIdx.x >> 5;
  __shared__ _Float16 Sld[4][16][32];

  v16h aq = load_a_f16(Qh + (size_t)(t0 + ln) * 32, g);     // loop-invariant
  const _Float16* vcol0 = Vth + (size_t)ln * 2048;
  const _Float16* vcol1 = Vth + (size_t)(16 + ln) * 2048;

  v16h kb0 = load_b_f16(Kh + (size_t)ln * 32, g);
  v16h kb1 = load_b_f16(Kh + (size_t)(16 + ln) * 32, g);
  v16h bv0 = load_b_f16(vcol0, g);
  v16h bv1 = load_b_f16(vcol1, g);
  v8f acc0 = {}, acc1 = {};
  for (int s0 = 0; s0 < 2048 - 32; s0 += 32) {
    v16h kn0 = load_b_f16(Kh + (size_t)(s0 + 32 + ln) * 32, g);
    v16h kn1 = load_b_f16(Kh + (size_t)(s0 + 48 + ln) * 32, g);
    v16h vn0 = load_b_f16(vcol0 + s0 + 32, g);
    v16h vn1 = load_b_f16(vcol1 + s0 + 32, g);
    attn_step(aq, kb0, kb1, bv0, bv1, Sld[ws], g, ln, acc0, acc1);
    kb0 = kn0; kb1 = kn1; bv0 = vn0; bv1 = vn1;
  }
  attn_step(aq, kb0, kb1, bv0, bv1, Sld[ws], g, ln, acc0, acc1);

  _Float16* Oh = A2 + (size_t)head * 65536;
#pragma unroll
  for (int j = 0; j < 8; ++j) {
    size_t base = (size_t)(t0 + j + 8 * g) * 32 + ln;
    Oh[base]      = (_Float16)acc0[j];
    Oh[base + 16] = (_Float16)acc1[j];
  }
}

// ---- Kernel 4: avg_weights[b] = (1/32) sum_head Q_h K_h^T ----------------
// 8192 waves: 2 batches x 128 t-tiles x 32 s-blocks(64). Pipelined over heads.
__global__ __launch_bounds__(128) void k_avg(
    const _Float16* __restrict__ Pq, const _Float16* __restrict__ Pk,
    float* __restrict__ avg)
{
  int wid = blockIdx.x * 4 + (threadIdx.x >> 5);
  int b   = wid >> 12;
  int r   = wid & 4095;
  int t0  = (r >> 5) << 4;
  int s0  = (r & 31) << 6;
  int lane = threadIdx.x & 31, g = lane >> 4, ln = lane & 15;
  const _Float16* Qb = Pq + (size_t)(b * 32) * 65536;
  const _Float16* Kb = Pk + (size_t)(b * 32) * 65536;
  v8f acc0 = {}, acc1 = {}, acc2 = {}, acc3 = {};

  v16h a  = load_a_f16(Qb + (size_t)(t0 + ln) * 32, g);
  v16h b0 = load_b_f16(Kb + (size_t)(s0 + ln) * 32, g);
  v16h b1 = load_b_f16(Kb + (size_t)(s0 + 16 + ln) * 32, g);
  v16h b2 = load_b_f16(Kb + (size_t)(s0 + 32 + ln) * 32, g);
  v16h b3 = load_b_f16(Kb + (size_t)(s0 + 48 + ln) * 32, g);
  for (int head = 1; head < 32; ++head) {
    const _Float16* Qh = Qb + (size_t)head * 65536;
    const _Float16* Kh = Kb + (size_t)head * 65536;
    v16h an  = load_a_f16(Qh + (size_t)(t0 + ln) * 32, g);
    v16h bn0 = load_b_f16(Kh + (size_t)(s0 + ln) * 32, g);
    v16h bn1 = load_b_f16(Kh + (size_t)(s0 + 16 + ln) * 32, g);
    v16h bn2 = load_b_f16(Kh + (size_t)(s0 + 32 + ln) * 32, g);
    v16h bn3 = load_b_f16(Kh + (size_t)(s0 + 48 + ln) * 32, g);
    acc0 = WMMA_F16(a, b0, acc0);
    acc1 = WMMA_F16(a, b1, acc1);
    acc2 = WMMA_F16(a, b2, acc2);
    acc3 = WMMA_F16(a, b3, acc3);
    a = an; b0 = bn0; b1 = bn1; b2 = bn2; b3 = bn3;
  }
  acc0 = WMMA_F16(a, b0, acc0);
  acc1 = WMMA_F16(a, b1, acc1);
  acc2 = WMMA_F16(a, b2, acc2);
  acc3 = WMMA_F16(a, b3, acc3);

  float* out = avg + (size_t)b * 2048 * 2048;
  const float sc = 1.0f / 32.0f;
#pragma unroll
  for (int j = 0; j < 8; ++j) {
    size_t row = (size_t)(t0 + j + 8 * g) * 2048 + s0 + ln;
    out[row]      = acc0[j] * sc;
    out[row + 16] = acc1[j] * sc;
    out[row + 32] = acc2[j] * sc;
    out[row + 48] = acc3[j] * sc;
  }
}

// ---- Kernel 5: output projection  out = A2 @ Wo^T + bo (fp32 out) --------
// 4096 waves: 256 row-tiles x 16 n-blocks(64). Pipelined over K-chunks.
__global__ __launch_bounds__(128) void k_oproj(
    const _Float16* __restrict__ A2, const _Float16* __restrict__ Wo,
    const float* __restrict__ bo, float* __restrict__ out)
{
  const int N = 1024, K = 1024;
  int wid = blockIdx.x * 4 + (threadIdx.x >> 5);
  int tm  = (wid >> 4) << 4;
  int tn  = (wid & 15) << 6;
  int lane = threadIdx.x & 31, g = lane >> 4, ln = lane & 15;
  v8f acc0 = {}, acc1 = {}, acc2 = {}, acc3 = {};
  const _Float16* arow = A2 + (size_t)(tm + ln) * K;
  const _Float16* wrow = Wo + (size_t)(tn + ln) * K;

  v16h a  = load_a_f16(arow, g);
  v16h b0 = load_b_f16(wrow, g);
  v16h b1 = load_b_f16(wrow + (size_t)16 * K, g);
  v16h b2 = load_b_f16(wrow + (size_t)32 * K, g);
  v16h b3 = load_b_f16(wrow + (size_t)48 * K, g);
  for (int k0 = 32; k0 < K; k0 += 32) {
    v16h an  = load_a_f16(arow + k0, g);
    v16h bn0 = load_b_f16(wrow + k0, g);
    v16h bn1 = load_b_f16(wrow + (size_t)16 * K + k0, g);
    v16h bn2 = load_b_f16(wrow + (size_t)32 * K + k0, g);
    v16h bn3 = load_b_f16(wrow + (size_t)48 * K + k0, g);
    acc0 = WMMA_F16(a, b0, acc0);
    acc1 = WMMA_F16(a, b1, acc1);
    acc2 = WMMA_F16(a, b2, acc2);
    acc3 = WMMA_F16(a, b3, acc3);
    a = an; b0 = bn0; b1 = bn1; b2 = bn2; b3 = bn3;
  }
  acc0 = WMMA_F16(a, b0, acc0);
  acc1 = WMMA_F16(a, b1, acc1);
  acc2 = WMMA_F16(a, b2, acc2);
  acc3 = WMMA_F16(a, b3, acc3);

  float bb0 = bo[tn + ln], bb1 = bo[tn + 16 + ln];
  float bb2 = bo[tn + 32 + ln], bb3 = bo[tn + 48 + ln];
#pragma unroll
  for (int j = 0; j < 8; ++j) {
    size_t base = (size_t)(tm + j + 8 * g) * N + tn + ln;
    out[base]      = acc0[j] + bb0;
    out[base + 16] = acc1[j] + bb1;
    out[base + 32] = acc2[j] + bb2;
    out[base + 48] = acc3[j] + bb3;
  }
}

extern "C" void kernel_launch(void* const* d_in, const int* in_sizes, int n_in,
                              void* d_out, int out_size, void* d_ws, size_t ws_size,
                              hipStream_t stream) {
  // inputs: query,key,value, Wq,bq, Wk,bk, Wv,bv, Wo,bo (all fp32)
  const float* q  = (const float*)d_in[0];
  const float* Wq = (const float*)d_in[3];
  const float* bq = (const float*)d_in[4];
  const float* Wk = (const float*)d_in[5];
  const float* bk = (const float*)d_in[6];
  const float* Wv = (const float*)d_in[7];
  const float* bv = (const float*)d_in[8];
  const float* Wo = (const float*)d_in[9];
  const float* bo = (const float*)d_in[10];

  float* out = (float*)d_out;            // attn: 2*2048*1024
  float* avg = out + (size_t)4194304;    // avg_weights: 2*2048*2048

  // workspace (halfs): Xh 4M | Wq/Wk/Wv/Wo 1M each | Pq/Pk/Pv 4M each
  //                    | Vt 4M | A2 4M  (56 MB total)
  _Float16* Xh  = (_Float16*)d_ws;
  _Float16* Wqh = Xh  + (size_t)4194304;
  _Float16* Wkh = Wqh + (size_t)1048576;
  _Float16* Wvh = Wkh + (size_t)1048576;
  _Float16* Woh = Wvh + (size_t)1048576;
  _Float16* Pq  = Woh + (size_t)1048576;
  _Float16* Pk  = Pq  + (size_t)4194304;
  _Float16* Pv  = Pk  + (size_t)4194304;
  _Float16* Vt  = Pv  + (size_t)4194304;
  _Float16* A2  = Vt  + (size_t)4194304;

  k_cvt<<<4096, 256, 0, stream>>>(q,  Xh,  1048576);
  k_cvt<<<1024, 256, 0, stream>>>(Wq, Wqh, 262144);
  k_cvt<<<1024, 256, 0, stream>>>(Wk, Wkh, 262144);
  k_cvt<<<1024, 256, 0, stream>>>(Wv, Wvh, 262144);
  k_cvt<<<1024, 256, 0, stream>>>(Wo, Woh, 262144);

  k_proj <<<3072, 128, 0, stream>>>(Xh, Wqh, bq, Pq, Wkh, bk, Pk, Wvh, bv, Pv);
  k_vt   <<<8192, 256, 0, stream>>>(Pv, Vt);
  k_attn <<<2048, 128, 0, stream>>>(Pq, Pk, Vt, A2);
  k_avg  <<<2048, 128, 0, stream>>>(Pq, Pk, avg);
  k_oproj<<<1024, 128, 0, stream>>>(A2, Woh, bo, out);
}
